// EntityExtractor_31679678775742
// MI455X (gfx1250) — compile-verified, matrix operand order checked by reading, not run on hardware
//
#include <hip/hip_runtime.h>

typedef __attribute__((ext_vector_type(8)))  _Float16 v8h;
typedef __attribute__((ext_vector_type(16))) _Float16 v16h;
typedef __attribute__((ext_vector_type(8)))  float    v8f;

// Build a 16-element f16 WMMA fragment from two contiguous 16B chunks.
// CDNA5 f16 A/B lane layout: lane<16 -> K in {base..base+7, base+16..base+23}
// with base=0; lanes 16..31 use base=8.  p points at the 32-element K row.
__device__ __forceinline__ v16h load_frag(const _Float16* p, int base) {
    v8h a = *(const v8h*)(p + base);
    v8h b = *(const v8h*)(p + base + 16);
    return __builtin_shufflevector(a, b, 0,1,2,3,4,5,6,7,8,9,10,11,12,13,14,15);
}

// ---------------------------------------------------------------------------
// Kernel 1: conv1 3x3 (3->32) + bias + ReLU + 2x2 maxpool, fp32 VALU.
// One thread per pooled pixel; emits NHWC f16 (512,512,32) for WMMA conv2.
// ---------------------------------------------------------------------------
__global__ __launch_bounds__(256) void conv1_pool_kernel(
    const float* __restrict__ obs, const float* __restrict__ w1,
    const float* __restrict__ b1, _Float16* __restrict__ act1) {
    int idx = blockIdx.x * 256 + threadIdx.x;       // 0 .. 512*512-1
    int px = idx & 511;
    int py = idx >> 9;

    float win[3][4][4];
#pragma unroll
    for (int ci = 0; ci < 3; ++ci)
#pragma unroll
        for (int i = 0; i < 4; ++i)
#pragma unroll
            for (int j = 0; j < 4; ++j) {
                int gy = 2 * py - 1 + i, gx = 2 * px - 1 + j;
                float v = 0.f;
                if (gy >= 0 && gy < 1024 && gx >= 0 && gx < 1024)
                    v = obs[ci * 1048576 + gy * 1024 + gx];
                win[ci][i][j] = v;
            }

    for (int op = 0; op < 16; ++op) {               // output-channel pairs
        unsigned pack = 0;
#pragma unroll
        for (int s = 0; s < 2; ++s) {
            int oc = op * 2 + s;
            float bv  = b1[oc];
            float a00 = bv, a01 = bv, a10 = bv, a11 = bv;
#pragma unroll
            for (int ci = 0; ci < 3; ++ci)
#pragma unroll
                for (int kh = 0; kh < 3; ++kh)
#pragma unroll
                    for (int kw = 0; kw < 3; ++kw) {
                        float wv = w1[((oc * 3 + ci) * 3 + kh) * 3 + kw];
                        a00 = fmaf(win[ci][kh    ][kw    ], wv, a00);
                        a01 = fmaf(win[ci][kh    ][kw + 1], wv, a01);
                        a10 = fmaf(win[ci][kh + 1][kw    ], wv, a10);
                        a11 = fmaf(win[ci][kh + 1][kw + 1], wv, a11);
                    }
            float m = fmaxf(fmaxf(a00, a01), fmaxf(a10, a11));
            m = fmaxf(m, 0.f);                      // relu(max)==max(relu)
            _Float16 h = (_Float16)m;
            unsigned short u;
            __builtin_memcpy(&u, &h, 2);
            pack |= ((unsigned)u) << (16 * s);
        }
        *(unsigned*)(&act1[(size_t)idx * 32 + op * 2]) = pack;
    }
}

// ---------------------------------------------------------------------------
// Kernel 2: conv2 3x3 (32->64) + bias + ReLU + 2x2 maxpool as implicit GEMM
// on v_wmma_f32_16x16x32_f16.  M=64 (4 tiles), K=9 taps * 32ci, N=16 px/wave.
// Grid: (4 x-strips of 128 cols, 256 pooled rows), 256 threads = 8 waves.
// Each wave: 16 conv cols, 2 conv rows -> 8 pooled px * 64 ch.  72 WMMAs.
// ---------------------------------------------------------------------------
__global__ __launch_bounds__(256) void conv2_pool_wmma_kernel(
    const _Float16* __restrict__ act1, const float* __restrict__ w2,
    const float* __restrict__ b2, _Float16* __restrict__ act2) {
    __shared__ __align__(16) _Float16 w2L[64 * 9 * 32];    // [oc][tap][ci] 36KB
    __shared__ __align__(16) _Float16 inL[4 * 130 * 32];   // [row][x][ci]  33KB
    __shared__ __align__(16) _Float16 poolL[8 * 2 * 16 * 64]; // [wave][r][n][ch]
    __shared__ float b2L[64];

    int tid   = threadIdx.x;
    int strip = blockIdx.x;                 // 0..3  (x-strip of 128 conv cols)
    int py    = blockIdx.y;                 // 0..255 (pooled row)
    int x0    = strip * 128;
    int y0    = 2 * py;

    // Repack weights (64,32,3,3) fp32 -> [oc][tap][ci] f16 in LDS.
    for (int i = tid; i < 64 * 288; i += 256) {
        int oc = i / 288, rem = i % 288;
        int tap = rem >> 5, ci = rem & 31;
        w2L[i] = (_Float16)w2[(oc * 32 + ci) * 9 + tap];
    }
    if (tid < 64) b2L[tid] = b2[tid];

    // Stage input tile: rows y0-1..y0+2, cols x0-1..x0+128, 32 ch, zero-pad.
    for (int i = tid; i < 4 * 130 * 32; i += 256) {
        int ci = i & 31, t2 = i >> 5;
        int x = t2 % 130, row = t2 / 130;
        int gx = x0 - 1 + x, gy = y0 - 1 + row;
        _Float16 v = (_Float16)0.f;
        if (gx >= 0 && gx < 512 && gy >= 0 && gy < 512)
            v = act1[((size_t)gy * 512 + gx) * 32 + ci];
        inL[i] = v;
    }
    __syncthreads();

    int lane = tid & 31, wave = tid >> 5;
    int base = (lane < 16) ? 0 : 8;
    int m15  = lane & 15;
    int wx   = wave * 16;                   // wave's conv-col offset in strip

    for (int r = 0; r < 2; ++r) {           // two conv rows feeding the pool
        v8f acc[4];
#pragma unroll
        for (int t = 0; t < 4; ++t)
#pragma unroll
            for (int v = 0; v < 8; ++v)
                acc[t][v] = b2L[t * 16 + v + base];   // bias pre-loaded into C

#pragma unroll
        for (int dy = 0; dy < 3; ++dy)
#pragma unroll
            for (int dx = 0; dx < 3; ++dx) {
                int tap = dy * 3 + dx;
                v16h bfrag = load_frag(&inL[((r + dy) * 130 + wx + m15 + dx) * 32], base);
#pragma unroll
                for (int t = 0; t < 4; ++t) {
                    v16h afrag = load_frag(&w2L[((t * 16 + m15) * 9 + tap) * 32], base);
                    acc[t] = __builtin_amdgcn_wmma_f32_16x16x32_f16(
                        false, afrag, false, bfrag, (short)0, acc[t], false, false);
                }
            }

        // ReLU + stash conv row into wave-local pool scratch [wave][r][n][ch].
#pragma unroll
        for (int t = 0; t < 4; ++t)
#pragma unroll
            for (int v = 0; v < 8; v += 2) {
                int ch = t * 16 + v + base;
                _Float16* dst = &poolL[(((wave * 2 + r) * 16) + m15) * 64 + ch];
                dst[0] = (_Float16)fmaxf(acc[t][v], 0.f);
                dst[1] = (_Float16)fmaxf(acc[t][v + 1], 0.f);
            }
    }
    __syncthreads();

    // 2x2 maxpool: lane -> (pooled px j, channel block cb of 16).
    int j  = lane >> 2;
    int cb = (lane & 3) << 4;
    v8h q[4][2];
#pragma unroll
    for (int k = 0; k < 4; ++k) {
        int rr = k >> 1, nn = 2 * j + (k & 1);
        const _Float16* src = &poolL[(((wave * 2 + rr) * 16) + nn) * 64 + cb];
        q[k][0] = *(const v8h*)(src);
        q[k][1] = *(const v8h*)(src + 8);
    }
    v8h o0, o1;
#pragma unroll
    for (int c = 0; c < 8; ++c) {
        o0[c] = (_Float16)fmaxf(fmaxf((float)q[0][0][c], (float)q[1][0][c]),
                                fmaxf((float)q[2][0][c], (float)q[3][0][c]));
        o1[c] = (_Float16)fmaxf(fmaxf((float)q[0][1][c], (float)q[1][1][c]),
                                fmaxf((float)q[2][1][c], (float)q[3][1][c]));
    }
    int pxg = strip * 64 + wave * 8 + j;            // pooled x, 0..255
    size_t ga = (((size_t)py * 256 + pxg) * 64 + cb);
    *(v8h*)(&act2[ga])     = o0;
    *(v8h*)(&act2[ga + 8]) = o1;
}

// ---------------------------------------------------------------------------
// Kernel 3: conv3 1x1 (64->16) + sigmoid + channel-index max, WMMA 16x16x32.
// One wave -> 16 pixels; A = w3 (16x64 f16 in LDS), B straight from global.
// ---------------------------------------------------------------------------
__global__ __launch_bounds__(256) void conv3_reduce_wmma_kernel(
    const _Float16* __restrict__ act2, const float* __restrict__ w3,
    const float* __restrict__ b3, float* __restrict__ out) {
    __shared__ __align__(16) _Float16 w3L[16 * 64];
    int tid = threadIdx.x;
    for (int i = tid; i < 16 * 64; i += 256)
        w3L[i] = (_Float16)w3[i];                    // (16,64,1,1) -> [m][ci]
    __syncthreads();

    int lane = tid & 31, wave = tid >> 5;
    int base = (lane < 16) ? 0 : 8;
    int m15  = lane & 15;
    size_t pix0 = ((size_t)blockIdx.x * 8 + wave) * 16;

    v8f acc;
#pragma unroll
    for (int v = 0; v < 8; ++v) acc[v] = b3[v + base];

#pragma unroll
    for (int ks = 0; ks < 2; ++ks) {                 // K = 64 -> two k-steps
        v16h afrag = load_frag(&w3L[m15 * 64 + ks * 32], base);
        v16h bfrag = load_frag(&act2[(pix0 + m15) * 64 + ks * 32], base);
        acc = __builtin_amdgcn_wmma_f32_16x16x32_f16(
            false, afrag, false, bfrag, (short)0, acc, false, false);
    }

    // blackboard = max_k (k if sigmoid(x_k) != 0 else 0)
    float cand = 0.f;
#pragma unroll
    for (int v = 0; v < 8; ++v) {
        float z = 1.f / (1.f + __expf(-acc[v]));
        float m = (float)(v + base);
        if (z != 0.f && m > cand) cand = m;
    }
    cand = fmaxf(cand, __shfl_xor(cand, 16, 32));    // combine M-halves (same N)
    if (lane < 16) out[pix0 + m15] = cand;
}

// ---------------------------------------------------------------------------
extern "C" void kernel_launch(void* const* d_in, const int* in_sizes, int n_in,
                              void* d_out, int out_size, void* d_ws, size_t ws_size,
                              hipStream_t stream) {
    const float* obs = (const float*)d_in[0];
    const float* w1  = (const float*)d_in[1];
    const float* b1  = (const float*)d_in[2];
    const float* w2  = (const float*)d_in[3];
    const float* b2  = (const float*)d_in[4];
    const float* w3  = (const float*)d_in[5];
    const float* b3  = (const float*)d_in[6];
    float* out = (float*)d_out;

    _Float16* act1 = (_Float16*)d_ws;                               // 512*512*32 f16
    _Float16* act2 = (_Float16*)((char*)d_ws + (size_t)512 * 512 * 32 * 2); // 256*256*64 f16

    conv1_pool_kernel<<<1024, 256, 0, stream>>>(obs, w1, b1, act1);
    conv2_pool_wmma_kernel<<<dim3(4, 256), 256, 0, stream>>>(act1, w2, b2, act2);
    conv3_reduce_wmma_kernel<<<512, 256, 0, stream>>>(act2, w3, b3, out);
}
